// GlobalGluOut_9904194585361
// MI455X (gfx1250) — matrix operand order repaired and checked by reading, not compiled, not run on hardware
//
#include <hip/hip_runtime.h>
#include <math.h>

// ---------------------------------------------------------------------------
// GlobalGluOut on MI455X (gfx1250):
//   kp1 = x_flat @ K1^T, kp2 = x_flat @ K2^T   (per-field 2048x128x4992 GEMMs)
//   p   = sigmoid(sum_e kp1 * x)               (block-local reduction)
//   out = x * (1 - p + kp2 * p)                (fused epilogue)
// bf16 WMMA (v_wmma_f32_16x16x32_bf16) with f32 accumulation.
// Software-pipelined: global loads for chunk k+1 fly while chunk k's WMMAs run;
// double-buffered LDS -> one workgroup barrier per K-chunk.
// f32->bf16 packing: v_cvt_pk_bf16_f32 if available, else 2x v_add + v_perm_b32
// (round-to-nearest, ties-away), else manual RNE.
// ---------------------------------------------------------------------------

#define FIELD_LEN 39
#define EMBED_DIM 128
#define INPUT_DIM (FIELD_LEN * EMBED_DIM) // 4992
#define BATCH 2048
#define KCHUNK 32
#define KITERS (INPUT_DIM / KCHUNK) // 156 (even -> unroll-by-2 safe)
#define LDS_STRIDE 40               // 32 bf16 payload + 8 pad (bank-conflict free)

typedef __attribute__((ext_vector_type(16))) __bf16 v16bf;
typedef __attribute__((ext_vector_type(2)))  __bf16 v2bf;
typedef __attribute__((ext_vector_type(8)))  float  v8f;
typedef __attribute__((ext_vector_type(4)))  float  f32x4;
typedef __attribute__((ext_vector_type(4)))  unsigned u32x4;

union FragBF {
    v16bf v;
    u32x4 q[2];
};

struct R8 { f32x4 a, b; }; // 8 in-flight f32 per matrix per thread

#if defined(__has_builtin)
#if __has_builtin(__builtin_amdgcn_cvt_pk_bf16_f32)
#define GLU_HW_PK_BF16 1
#elif __has_builtin(__builtin_amdgcn_perm)
#define GLU_PERM_PK_BF16 1
#endif
#endif

__device__ __forceinline__ unsigned pk2(float lo, float hi) {
#if defined(GLU_HW_PK_BF16)
    // Single v_cvt_pk_bf16_f32 (hardware RNE).
    v2bf v = __builtin_amdgcn_cvt_pk_bf16_f32(lo, hi);
    return __builtin_bit_cast(unsigned, v);
#elif defined(GLU_PERM_PK_BF16)
    // Round-to-nearest (ties away): one v_add_nc_u32 per float (VOPD-pairable),
    // then a single v_perm_b32 grabs both high halves:
    //   sel 0x07060302 -> D = { uh[31:24], uh[23:16], ul[31:24], ul[23:16] }
    unsigned ul = __float_as_uint(lo) + 0x8000u;
    unsigned uh = __float_as_uint(hi) + 0x8000u;
    return __builtin_amdgcn_perm(uh, ul, 0x07060302u);
#else
    // Manual round-to-nearest-even fallback.
    unsigned ul = __float_as_uint(lo);
    unsigned uh = __float_as_uint(hi);
    ul = ul + 0x7FFFu + ((ul >> 16) & 1u);
    uh = uh + 0x7FFFu + ((uh >> 16) & 1u);
    return (ul >> 16) | (uh & 0xFFFF0000u);
#endif
}

__device__ __forceinline__ R8 load8(const float* __restrict__ g) {
    R8 r;
    r.a = ((const f32x4*)g)[0];
    r.b = ((const f32x4*)g)[1];
    return r;
}
__device__ __forceinline__ void cvt_store8(__bf16* s, const R8& r) {
    u32x4 q = { pk2(r.a.x, r.a.y), pk2(r.a.z, r.a.w),
                pk2(r.b.x, r.b.y), pk2(r.b.z, r.b.w) };
    *(u32x4*)s = q;
}

// A-matrix fragment (16x32 bf16). ISA layout: lane L<16 -> row L, K 0..7 in
// v0-3 and K 16..23 in v4-7; lane L>=16 -> row L-16, K 8..15 / 24..31.
__device__ __forceinline__ v16bf lds_frag_a(const __bf16* tile, int lane) {
    const int r = lane & 15, g = lane >> 4;
    const __bf16* p = tile + r * LDS_STRIDE;
    FragBF f;
    f.q[0] = *(const u32x4*)(p + g * 8);
    f.q[1] = *(const u32x4*)(p + 16 + g * 8);
    return f.v;
}

// B-matrix fragment (32x16 bf16), stored as B^T (N-row x K-col) in LDS.
// ISA layout: lane L -> col L&15; K 0..7 / 16..23 (lanes 0-15), K 8..15 / 24..31 (lanes 16-31).
__device__ __forceinline__ v16bf lds_frag_b(const __bf16* tile, int lane) {
    const int r = lane & 15, g = lane >> 4;
    const __bf16* p = tile + r * LDS_STRIDE;
    FragBF f;
    f.q[0] = *(const u32x4*)(p + g * 8);
    f.q[1] = *(const u32x4*)(p + 16 + g * 8);
    return f.v;
}

__global__ __launch_bounds__(512, 1) void glu_wmma_kernel(
    const float* __restrict__ x,   // (2048, 39, 128) == x_flat (2048, 4992)
    const float* __restrict__ k1,  // (128, 39, 4992)
    const float* __restrict__ k2,  // (128, 39, 4992)
    float* __restrict__ out)       // (2048, 39, 128)
{
    __shared__ __bf16 sA [2][128 * LDS_STRIDE];
    __shared__ __bf16 sB1[2][128 * LDS_STRIDE];
    __shared__ __bf16 sB2[2][128 * LDS_STRIDE];
    __shared__ float  s_red[128];

    const int tid  = threadIdx.x;
    const int lane = tid & 31;
    const int w    = tid >> 5;   // 16 wave32s
    const int wm   = w & 3;      // row group: 4 x 32 rows
    const int wn   = w >> 2;     // col group: 4 x 32 cols
    const int bm   = blockIdx.x; // batch tile (128 rows)
    const int f    = blockIdx.y; // field

    // Cooperative load mapping: 512 threads cover 128 rows x 32 K, 8 f32 each.
    const int lr = tid >> 2;         // 0..127
    const int lc = (tid & 3) << 3;   // 0, 8, 16, 24

    const float* gA  = x  + (size_t)(bm * 128 + lr) * INPUT_DIM + lc;
    const float* gB1 = k1 + ((size_t)lr * FIELD_LEN + f) * INPUT_DIM + lc;
    const float* gB2 = k2 + ((size_t)lr * FIELD_LEN + f) * INPUT_DIM + lc;
    const int sOff = lr * LDS_STRIDE + lc;

    v8f acc1[2][2] = {};
    v8f acc2[2][2] = {};

    // Prologue: chunk 0 in flight.
    R8 rA  = load8(gA);
    R8 rB1 = load8(gB1);
    R8 rB2 = load8(gB2);

#pragma unroll 2
    for (int kc = 0; kc < KITERS; ++kc) {
        const int buf = kc & 1;
        __bf16* sAb  = &sA [buf][0];
        __bf16* sB1b = &sB1[buf][0];
        __bf16* sB2b = &sB2[buf][0];

        // Drain in-flight chunk k into LDS[buf] (waits loadcnt here).
        cvt_store8(sAb  + sOff, rA);
        cvt_store8(sB1b + sOff, rB1);
        cvt_store8(sB2b + sOff, rB2);

        // Launch chunk k+1 global loads; they fly during chunk k's WMMAs.
        if (kc + 1 < KITERS) { // uniform branch
            const int k0 = (kc + 1) * KCHUNK;
            rA  = load8(gA  + k0);
            rB1 = load8(gB1 + k0);
            rB2 = load8(gB2 + k0);
            __builtin_prefetch(gA  + k0 + KCHUNK, 0, 0); // global_prefetch_b8
            __builtin_prefetch(gB1 + k0 + KCHUNK, 0, 0);
            __builtin_prefetch(gB2 + k0 + KCHUNK, 0, 0);
        }

        __syncthreads(); // LDS[buf] tiles visible workgroup-wide

        const v16bf a0 = lds_frag_a(sAb + (wm * 32 + 0)  * LDS_STRIDE, lane);
        const v16bf a1 = lds_frag_a(sAb + (wm * 32 + 16) * LDS_STRIDE, lane);

        const v16bf b10 = lds_frag_b(sB1b + (wn * 32 + 0)  * LDS_STRIDE, lane);
        const v16bf b11 = lds_frag_b(sB1b + (wn * 32 + 16) * LDS_STRIDE, lane);
        const v16bf b20 = lds_frag_b(sB2b + (wn * 32 + 0)  * LDS_STRIDE, lane);
        const v16bf b21 = lds_frag_b(sB2b + (wn * 32 + 16) * LDS_STRIDE, lane);

        acc1[0][0] = __builtin_amdgcn_wmma_f32_16x16x32_bf16(
            false, a0, false, b10, (short)0, acc1[0][0], false, false);
        acc1[1][0] = __builtin_amdgcn_wmma_f32_16x16x32_bf16(
            false, a1, false, b10, (short)0, acc1[1][0], false, false);
        acc1[0][1] = __builtin_amdgcn_wmma_f32_16x16x32_bf16(
            false, a0, false, b11, (short)0, acc1[0][1], false, false);
        acc1[1][1] = __builtin_amdgcn_wmma_f32_16x16x32_bf16(
            false, a1, false, b11, (short)0, acc1[1][1], false, false);

        acc2[0][0] = __builtin_amdgcn_wmma_f32_16x16x32_bf16(
            false, a0, false, b20, (short)0, acc2[0][0], false, false);
        acc2[1][0] = __builtin_amdgcn_wmma_f32_16x16x32_bf16(
            false, a1, false, b20, (short)0, acc2[1][0], false, false);
        acc2[0][1] = __builtin_amdgcn_wmma_f32_16x16x32_bf16(
            false, a0, false, b21, (short)0, acc2[0][1], false, false);
        acc2[1][1] = __builtin_amdgcn_wmma_f32_16x16x32_bf16(
            false, a1, false, b21, (short)0, acc2[1][1], false, false);
    }

    // -------- fused epilogue: p = sigmoid(sum_e kp1*x); out = x*(1-p+kp2*p) --
    // C/D layout: v8f element i of a 16x16 tile lives at row (i + 8*(lane>>4)),
    // col (lane&15).
    const int g  = lane >> 4;
    const int cl = lane & 15;

    __syncthreads();
    if (tid < 128) s_red[tid] = 0.0f;
    __syncthreads();

#pragma unroll
    for (int mt = 0; mt < 2; ++mt) {
        float part[8];
#pragma unroll
        for (int i = 0; i < 8; ++i) part[i] = 0.0f;
#pragma unroll
        for (int nt = 0; nt < 2; ++nt) {
            const int e = wn * 32 + nt * 16 + cl;
#pragma unroll
            for (int i = 0; i < 8; ++i) {
                const int ml = wm * 32 + mt * 16 + i + 8 * g;
                const size_t b = (size_t)(bm * 128 + ml);
                const float xv = x[b * INPUT_DIM + f * EMBED_DIM + e];
                part[i] += acc1[mt][nt][i] * xv;
            }
        }
#pragma unroll
        for (int i = 0; i < 8; ++i) {
            float v = part[i];
            v += __shfl_xor(v, 1, 32);
            v += __shfl_xor(v, 2, 32);
            v += __shfl_xor(v, 4, 32);
            v += __shfl_xor(v, 8, 32); // summed within each 16-lane half
            if (cl == 0) {
                const int ml = wm * 32 + mt * 16 + i + 8 * g;
                atomicAdd(&s_red[ml], v); // four wn-groups contribute per row
            }
        }
    }
    __syncthreads();
    if (tid < 128) {
        const float s = s_red[tid];
        s_red[tid] = 1.0f / (1.0f + __expf(-s));
    }
    __syncthreads();

#pragma unroll
    for (int mt = 0; mt < 2; ++mt) {
#pragma unroll
        for (int nt = 0; nt < 2; ++nt) {
            const int e = wn * 32 + nt * 16 + cl;
#pragma unroll
            for (int i = 0; i < 8; ++i) {
                const int ml = wm * 32 + mt * 16 + i + 8 * g;
                const size_t b = (size_t)(bm * 128 + ml);
                const float xv = x[b * INPUT_DIM + f * EMBED_DIM + e];
                const float p  = s_red[ml];
                out[b * INPUT_DIM + f * EMBED_DIM + e] =
                    xv * (1.0f - p + acc2[mt][nt][i] * p);
            }
        }
    }
}

extern "C" void kernel_launch(void* const* d_in, const int* in_sizes, int n_in,
                              void* d_out, int out_size, void* d_ws, size_t ws_size,
                              hipStream_t stream) {
    (void)in_sizes; (void)n_in; (void)out_size; (void)d_ws; (void)ws_size;
    const float* x  = (const float*)d_in[0];
    const float* k1 = (const float*)d_in[1];
    const float* k2 = (const float*)d_in[2];
    float* out = (float*)d_out;

    dim3 grid(BATCH / 128, FIELD_LEN); // f-major scheduling keeps x L2-resident
    glu_wmma_kernel<<<grid, 512, 0, stream>>>(x, k1, k2, out);
}
